// LaGraphNetNode_73830487818445
// MI455X (gfx1250) — matrix operand order repaired and checked by reading, not compiled, not run on hardware
//
#include <hip/hip_runtime.h>
#include <hip/hip_bf16.h>
#include <stdint.h>
#include <stddef.h>

// Problem constants (reference: N=8192, D=128, L=5)
#define N_NODES 8192
#define DFEAT   128
#define NL      5
#define BN_EPS  1e-5f

typedef __bf16 bf16;
typedef bf16  bf16x8  __attribute__((ext_vector_type(8)));
typedef bf16  bf16x16 __attribute__((ext_vector_type(16)));
typedef float f32x8   __attribute__((ext_vector_type(8)));
typedef float f32x4   __attribute__((ext_vector_type(4)));   // native vector (not HIP struct)

union Frag { bf16x16 v; struct { bf16x8 lo, hi; } p; };

#if defined(__has_builtin)
#  if __has_builtin(__builtin_amdgcn_sched_barrier)
#    define SCHED_FENCE() __builtin_amdgcn_sched_barrier(0)
#  endif
#endif
#ifndef SCHED_FENCE
#  define SCHED_FENCE()
#endif

// ---------------------------------------------------------------------------
// fp32 -> bf16 bulk convert (8 elems/thread, b128 in / b128 out, non-temporal:
// the 256MB adj read and 128MB adjB write are pure streaming traffic)
// ---------------------------------------------------------------------------
__global__ void cvt_f32_bf16(const float* __restrict__ in, bf16* __restrict__ out,
                             size_t n8) {
  size_t i = (size_t)blockIdx.x * blockDim.x + threadIdx.x;
  if (i >= n8) return;
  const f32x4* p = (const f32x4*)in + i * 2;
  f32x4 a = __builtin_nontemporal_load(p);
  f32x4 b = __builtin_nontemporal_load(p + 1);
  bf16x8 o;
  o[0] = (bf16)a[0]; o[1] = (bf16)a[1]; o[2] = (bf16)a[2]; o[3] = (bf16)a[3];
  o[4] = (bf16)b[0]; o[5] = (bf16)b[1]; o[6] = (bf16)b[2]; o[7] = (bf16)b[3];
  __builtin_nontemporal_store(o, (bf16x8*)out + i);
}

// W[l][k][n] (fp32) -> WT[l][n][k] (bf16) : tiny, once per call
__global__ void cvt_w_transpose(const float* __restrict__ W, bf16* __restrict__ WT,
                                int total) {
  int i = blockIdx.x * blockDim.x + threadIdx.x;
  if (i >= total) return;
  int l = i >> 14, rem = i & 16383;
  int n = rem >> 7, k = rem & 127;
  WT[i] = (bf16)W[(l << 14) + (k << 7) + n];
}

// x_m = x*(1-m) + noise*m ; also emits x_m and mask to d_out, bf16 x_m for GEMM
__global__ void mask_kernel(const float* __restrict__ x, const float* __restrict__ mask,
                            const float* __restrict__ noise,
                            float* __restrict__ out_xm, float* __restrict__ out_mask,
                            bf16* __restrict__ hb, int n) {
  int i = blockIdx.x * blockDim.x + threadIdx.x;
  if (i >= n) return;
  float m = mask[i];
  float xm = x[i] * (1.0f - m) + noise[i] * m;
  out_xm[i] = xm;
  out_mask[i] = m;
  hb[i] = (bf16)xm;
}

// ---------------------------------------------------------------------------
// WMMA GEMM:  C[M x 128] = A[M x K](bf16, row-major) * B[K x 128]
// B supplied K-contiguous per column: BT[n][k], row length K.
// One wave computes a 16x128 strip (8 f32 accumulators); 4 waves/WG.
// B tile (32x128 = 8KB) staged per WG in LDS, double buffered, in
// fragment-major layout: [tile t][lane][16 elems] -> two sequential
// ds_read_b128 per fragment, zero conflicts.
// Pipeline per K-step: issue next-tile global loads -> 8 WMMAs off LDS ->
// commit staged regs to other LDS buffer -> barrier.
// MODE 0: small GEMM (h @ W) -> store bf16 TRANSPOSED into outT[128][8192]
// MODE 1: big GEMM (adj @ hW) -> +bias, PReLU, fp32 row-major out
// ---------------------------------------------------------------------------
template <int MODE>
__global__ __launch_bounds__(128) void gemm_bf16(
    const bf16* __restrict__ A, const bf16* __restrict__ BT, int K,
    const float* __restrict__ bias, const float* __restrict__ alphaPtr,
    float* __restrict__ outF, bf16* __restrict__ outT) {
  __shared__ __align__(16) bf16 ldsB[2][4096];   // 2 x 8KB

  const int tid  = threadIdx.x;
  const int lane = tid & 31;
  const int wave = tid >> 5;
  const int row  = lane & 15;   // A: row-in-tile, B/C: col-in-tile
  const int sel  = lane >> 4;   // lane half: K sub-range / C row half
  const int m0   = (blockIdx.x * 4 + wave) * 16;

  // A fragment source: 8 contiguous K at sel*8, +16 elems for the hi half
  const bf16* arow = A + (size_t)(m0 + row) * K + sel * 8;

  // Staging descriptors: 512 chunks of 16B; thread tid owns chunks tid*4+i.
  // chunk cid -> tile t=cid>>6, frag-lane ln=(cid&63)>>1, half=cid&1
  // src element = BT[(t*16 + (ln&15))*K + k0 + (ln>>4)*8 + half*16]
  // dst element = cid*8  (fragment-major: t*512 + ln*16 + half*8)
  size_t srcOff[4];
  int dstOff[4];
#pragma unroll
  for (int i = 0; i < 4; ++i) {
    int cid = tid * 4 + i;
    int t = cid >> 6, r = cid & 63, ln = r >> 1, half = r & 1;
    int col = (t << 4) + (ln & 15);
    int k = ((ln >> 4) << 3) + (half << 4);
    srcOff[i] = (size_t)col * K + k;
    dstOff[i] = cid << 3;
  }

  const f32x8 zero = {0.f, 0.f, 0.f, 0.f, 0.f, 0.f, 0.f, 0.f};
  f32x8 acc[8];
#pragma unroll
  for (int t = 0; t < 8; ++t) acc[t] = zero;

  // Prologue: stage B tile for k0=0 and prefetch A fragment 0
  bf16x8 st[4];
#pragma unroll
  for (int i = 0; i < 4; ++i) st[i] = *(const bf16x8*)(BT + srcOff[i]);
  Frag aN;
  aN.p.lo = __builtin_nontemporal_load((const bf16x8*)(arow));
  aN.p.hi = __builtin_nontemporal_load((const bf16x8*)(arow + 16));
#pragma unroll
  for (int i = 0; i < 4; ++i) *(bf16x8*)(&ldsB[0][dstOff[i]]) = st[i];
  __syncthreads();

  for (int k0 = 0; k0 < K; k0 += 32) {
    const int cur = (k0 >> 5) & 1;
    const bool hasNext = (k0 + 32) < K;
    Frag aC = aN;

    // Phase 1: issue long-latency global loads for the NEXT step
    if (hasNext) {
#pragma unroll
      for (int i = 0; i < 4; ++i)
        st[i] = *(const bf16x8*)(BT + srcOff[i] + (k0 + 32));
      aN.p.lo = __builtin_nontemporal_load((const bf16x8*)(arow + k0 + 32));
      aN.p.hi = __builtin_nontemporal_load((const bf16x8*)(arow + k0 + 48));
    }
    SCHED_FENCE();

    // Phase 2: 8 WMMAs off the current LDS buffer (short-latency ds reads)
    const bf16* lb = &ldsB[cur][0];
#pragma unroll
    for (int t = 0; t < 8; ++t) {
      const bf16* p = lb + t * 512 + lane * 16;
      Frag b;
      b.p.lo = *(const bf16x8*)p;
      b.p.hi = *(const bf16x8*)(p + 8);
      acc[t] = __builtin_amdgcn_wmma_f32_16x16x32_bf16(
          false, aC.v, false, b.v, (short)0, acc[t], false, false);
    }
    SCHED_FENCE();

    // Phase 3: commit staged registers to the other LDS buffer
    if (hasNext) {
#pragma unroll
      for (int i = 0; i < 4; ++i) *(bf16x8*)(&ldsB[cur ^ 1][dstOff[i]]) = st[i];
    }
    __syncthreads();
  }

  if constexpr (MODE == 1) {
    const float alpha = alphaPtr[0];
#pragma unroll
    for (int t = 0; t < 8; ++t) {
      const int col = t * 16 + row;
      const float bc = bias[col];
      size_t base = (size_t)(m0 + sel * 8) * DFEAT + col;
#pragma unroll
      for (int r = 0; r < 8; ++r) {
        float v = acc[t][r] + bc;
        v = (v >= 0.f) ? v : alpha * v;        // PReLU
        outF[base + (size_t)r * DFEAT] = v;
      }
    }
  } else {
    // store transposed bf16: outT[col][m], 8 consecutive rows per lane -> b128
#pragma unroll
    for (int t = 0; t < 8; ++t) {
      const int col = t * 16 + row;
      bf16x8 pk;
#pragma unroll
      for (int r = 0; r < 8; ++r) pk[r] = (bf16)acc[t][r];
      *(bf16x8*)(outT + (size_t)col * N_NODES + m0 + sel * 8) = pk;
    }
  }
}

// ---------------------------------------------------------------------------
// BatchNorm over node axis (training mode, biased variance)
// ---------------------------------------------------------------------------
__global__ void bn_zero(float* __restrict__ s, float* __restrict__ q) {
  int t = threadIdx.x;
  if (t < DFEAT) { s[t] = 0.f; q[t] = 0.f; }
}

__global__ __launch_bounds__(256) void bn_stats(const float* __restrict__ x,
                                                float* __restrict__ gsum,
                                                float* __restrict__ gsq) {
  __shared__ float ss[256], sq[256];
  const int t = threadIdx.x;
  const int col = t & 127;
  const int half = t >> 7;
  const int r0 = blockIdx.x * 32;          // 256 blocks x 32 rows = 8192
  float s = 0.f, q = 0.f;
  for (int rr = half; rr < 32; rr += 2) {
    float v = x[(size_t)(r0 + rr) * DFEAT + col];
    s += v; q += v * v;
  }
  ss[t] = s; sq[t] = q;
  __syncthreads();
  if (t < 128) {
    atomicAdd(&gsum[col], ss[t] + ss[t + 128]);
    atomicAdd(&gsq[col],  sq[t] + sq[t + 128]);
  }
}

__global__ void bn_apply(const float* __restrict__ x,
                         const float* __restrict__ gsum, const float* __restrict__ gsq,
                         const float* __restrict__ gamma, const float* __restrict__ beta,
                         float* __restrict__ yF, bf16* __restrict__ yB, int n) {
  int i = blockIdx.x * blockDim.x + threadIdx.x;
  if (i >= n) return;
  int col = i & 127;
  const float invN = 1.0f / (float)N_NODES;
  float mu  = gsum[col] * invN;
  float var = gsq[col] * invN - mu * mu;
  float inv = rsqrtf(var + BN_EPS);
  float y = gamma[col] * ((x[i] - mu) * inv) + beta[col];
  yF[i] = y;
  yB[i] = (bf16)y;
}

// ---------------------------------------------------------------------------
// Host orchestration
// ---------------------------------------------------------------------------
extern "C" void kernel_launch(void* const* d_in, const int* in_sizes, int n_in,
                              void* d_out, int out_size, void* d_ws, size_t ws_size,
                              hipStream_t stream) {
  const float* x      = (const float*)d_in[0];
  const float* adj    = (const float*)d_in[1];
  const float* mask   = (const float*)d_in[2];
  const float* noise  = (const float*)d_in[3];
  const float* enc_W  = (const float*)d_in[4];
  const float* enc_b  = (const float*)d_in[5];
  const float* enc_a  = (const float*)d_in[6];
  const float* enc_g  = (const float*)d_in[7];
  const float* enc_be = (const float*)d_in[8];
  const float* dec_W  = (const float*)d_in[9];
  const float* dec_b  = (const float*)d_in[10];
  const float* dec_a  = (const float*)d_in[11];
  const float* dec_g  = (const float*)d_in[12];
  const float* dec_be = (const float*)d_in[13];

  const size_t ND = (size_t)N_NODES * DFEAT;      // 1,048,576
  float* out_xm   = (float*)d_out;                // section 0: masked x
  float* out_enc  = (float*)d_out + ND;           // section 1: encoded
  float* out_dec  = (float*)d_out + 2 * ND;       // section 2: decoded
  float* out_mask = (float*)d_out + 3 * ND;       // section 3: mask copy

  // Workspace carve-out (256B aligned)
  size_t off = 0;
  auto carve = [&](size_t bytes) {
    void* p = (char*)d_ws + off;
    off = (off + bytes + 255) & ~(size_t)255;
    return p;
  };
  bf16*  adjB  = (bf16*)carve((size_t)N_NODES * N_NODES * sizeof(bf16)); // 128 MB
  bf16*  tbT   = (bf16*)carve(ND * sizeof(bf16));                        // hW transposed
  bf16*  hb    = (bf16*)carve(ND * sizeof(bf16));                        // activations bf16
  float* gout  = (float*)carve(ND * sizeof(float));                      // GCN out fp32
  float* hfp   = (float*)carve(ND * sizeof(float));                      // BN fp32 scratch
  bf16*  WTb   = (bf16*)carve((size_t)2 * NL * DFEAT * DFEAT * sizeof(bf16));
  float* bnsum = (float*)carve(DFEAT * sizeof(float));
  float* bnsq  = (float*)carve(DFEAT * sizeof(float));
  (void)ws_size; (void)n_in; (void)in_sizes; (void)out_size;

  // 1) adj fp32 -> bf16 (streamed once; halves per-layer HBM traffic)
  {
    size_t n8 = (size_t)N_NODES * N_NODES / 8;
    cvt_f32_bf16<<<(unsigned)((n8 + 255) / 256), 256, 0, stream>>>(adj, adjB, n8);
  }
  // 2) weights -> bf16, transposed to K-contiguous-per-column layout
  cvt_w_transpose<<<(NL * DFEAT * DFEAT + 255) / 256, 256, 0, stream>>>(
      enc_W, WTb, NL * DFEAT * DFEAT);
  cvt_w_transpose<<<(NL * DFEAT * DFEAT + 255) / 256, 256, 0, stream>>>(
      dec_W, WTb + (size_t)NL * DFEAT * DFEAT, NL * DFEAT * DFEAT);
  // 3) masking (+ emit x_m, mask outputs, bf16 input activations)
  mask_kernel<<<(unsigned)(ND / 256), 256, 0, stream>>>(x, mask, noise, out_xm,
                                                        out_mask, hb, (int)ND);

  // 4) 10 GCN layers: encoder (BN after each), decoder (BN except last)
  for (int li = 0; li < 2 * NL; ++li) {
    const bool isEnc = li < NL;
    const int s = isEnc ? li : li - NL;
    const float* b_  = (isEnc ? enc_b  : dec_b)  + s * DFEAT;
    const float* al_ = (isEnc ? enc_a  : dec_a)  + s;
    const float* g_  = (isEnc ? enc_g  : dec_g)  + s * DFEAT;
    const float* be_ = (isEnc ? enc_be : dec_be) + s * DFEAT;
    const bf16* Wt = WTb + (size_t)li * DFEAT * DFEAT;

    // t = h @ W  (stored transposed bf16 for the big GEMM's B operand)
    gemm_bf16<0><<<N_NODES / 64, 128, 0, stream>>>(hb, Wt, DFEAT, nullptr, nullptr,
                                                   nullptr, tbT);
    // out = PReLU(adj @ t + b)
    float* gdst = (li == 2 * NL - 1) ? out_dec : gout;
    gemm_bf16<1><<<N_NODES / 64, 128, 0, stream>>>(adjB, tbT, N_NODES, b_, al_,
                                                   gdst, nullptr);
    if (li != 2 * NL - 1) {
      bn_zero<<<1, 128, 0, stream>>>(bnsum, bnsq);
      bn_stats<<<N_NODES / 32, 256, 0, stream>>>(gout, bnsum, bnsq);
      float* hf = (li == NL - 1) ? out_enc : hfp;  // encoder output captured here
      bn_apply<<<(unsigned)(ND / 256), 256, 0, stream>>>(gout, bnsum, bnsq, g_, be_,
                                                         hf, hb, (int)ND);
    }
  }
}
